// DGCNNLanPointNet_80178449481752
// MI455X (gfx1250) — compile-verified, matrix operand order checked by reading, not run on hardware
//
#include <hip/hip_runtime.h>
#include <hip/hip_bf16.h>
#include <cstdint>
#include <cstddef>

typedef __attribute__((ext_vector_type(16))) _Float16 v16h;
typedef __attribute__((ext_vector_type(8)))  float    v8f;

constexpr int kB  = 8;       // batch
constexpr int kN  = 52;      // objects per batch
constexpr int kP  = 1024;    // points per cloud
constexpr int kNC = kB * kN; // 416 clouds
constexpr int kL  = 24;      // language length
constexpr int kD  = 128;     // feature dim

// ---------------------------------------------------------------------------
// small device helpers
// ---------------------------------------------------------------------------
__device__ __forceinline__ void lds_fence() {
  // CDNA5 split wait counter: make this wave's LDS stores visible to its lanes
  asm volatile("s_wait_dscnt 0x0" ::: "memory");
}

__device__ __forceinline__ float sigm(float x) { return 1.f / (1.f + __expf(-x)); }

// Convert two contiguous 8-float runs (p0[0..7], p1[0..7]) into one f16 A/B
// fragment (vectorized: 4x global_load_b128 / ds_load_b128, zero branches).
template <bool RELU>
__device__ __forceinline__ v16h cvt_frag(const float* __restrict__ p0,
                                         const float* __restrict__ p1)
{
  const float4 a = *(const float4*)p0;
  const float4 b = *(const float4*)(p0 + 4);
  const float4 c = *(const float4*)p1;
  const float4 d = *(const float4*)(p1 + 4);
  v16h r;
  auto cv = [](float v) -> _Float16 {
    if (RELU) v = v > 0.f ? v : 0.f;
    return (_Float16)v;
  };
  r[0]=cv(a.x);  r[1]=cv(a.y);  r[2]=cv(a.z);  r[3]=cv(a.w);
  r[4]=cv(b.x);  r[5]=cv(b.y);  r[6]=cv(b.z);  r[7]=cv(b.w);
  r[8]=cv(c.x);  r[9]=cv(c.y);  r[10]=cv(c.z); r[11]=cv(c.w);
  r[12]=cv(d.x); r[13]=cv(d.y); r[14]=cv(d.z); r[15]=cv(d.w);
  return r;
}

// ---------------------------------------------------------------------------
// generic WMMA GEMM:  C[M,N] = act( opA(A)[M,Kpad] @ Wpacked + bias )
// Requirements: M % 16 == 0; lda == Kpad (mult of 32, pad cols zeroed by the
// producer); W packed fragment-major: [ntile][ktile][lane][16] f16.
// AMODE 0: plain A; 1: relu(A); 2: virtual pair rows for the `u` tensor:
//   row r -> (i=r/416, j=(r/8)%52, b=r%8), A[r,k]=relu(k<128?G[b,i,k]:G[b,j,k-128])
// act: 0 none, 1 relu, 2 leaky(0.2)
// ---------------------------------------------------------------------------
template <int AMODE>
__global__ __launch_bounds__(256) void gemm_kernel(
    const float* __restrict__ A, int lda,
    const _Float16* __restrict__ W,
    const float* __restrict__ bias,
    float* __restrict__ C, int ldc,
    int M, int N, int K, int act,
    const float* __restrict__ G)
{
  const int lane = threadIdx.x & 31;
  const int wid  = threadIdx.x >> 5;
  const int wpb  = blockDim.x >> 5;
  const int half = lane >> 4;
  const int l16  = lane & 15;
  const int tilesN = (N + 15) >> 4;
  const int tilesM = M >> 4;            // M is always a multiple of 16
  const int total  = tilesM * tilesN;
  const int Kp     = (K + 31) & ~31;
  const int kTiles = Kp >> 5;

  for (int t = blockIdx.x * wpb + wid; t < total; t += gridDim.x * wpb) {
    const int tn = t % tilesN, tm = t / tilesN;
    const int rowBase = tm << 4, colBase = tn << 4;
    const int arow = rowBase + l16;

    const float* rowPtr = nullptr;
    const float* Gi = nullptr;
    const float* Gj = nullptr;
    if (AMODE == 2) {
      const int bb = arow & 7;
      const int ij = arow >> 3;
      Gi = G + (size_t)(bb * kN + ij / kN) * kD;
      Gj = G + (size_t)(bb * kN + ij % kN) * kD;
    } else {
      rowPtr = A + (size_t)arow * lda;
    }
    const _Float16* wp = W + ((size_t)tn * kTiles * 32 + lane) * 16;

    v8f acc = {};
    for (int kb = 0; kb < kTiles; ++kb) {
      const int base = (kb << 5) + (half << 3);
      v16h afrag;
      if (AMODE == 2) {
        const float* s0 = (base < kD)      ? (Gi + base)      : (Gj + base - kD);
        const float* s1 = (base + 16 < kD) ? (Gi + base + 16) : (Gj + base + 16 - kD);
        afrag = cvt_frag<true>(s0, s1);
      } else if (AMODE == 1) {
        afrag = cvt_frag<true>(rowPtr + base, rowPtr + base + 16);
      } else {
        afrag = cvt_frag<false>(rowPtr + base, rowPtr + base + 16);
      }
      const v16h bfrag = *(const v16h*)(wp + (size_t)kb * 512);
      acc = __builtin_amdgcn_wmma_f32_16x16x32_f16(false, afrag, false, bfrag,
                                                   (short)0, acc, false, false);
    }
#pragma unroll
    for (int r = 0; r < 8; ++r) {
      const int rr = rowBase + r + (half << 3);
      const int cc = colBase + l16;
      if (cc < N) {
        float v = acc[r] + bias[cc];
        if (act == 1)      v = v > 0.f ? v : 0.f;
        else if (act == 2) v = v > 0.f ? v : 0.2f * v;
        C[(size_t)rr * ldc + cc] = v;
      }
    }
  }
}

// ---------------------------------------------------------------------------
// weight fp32 [K,N] -> f16 packed fragment-major [ntile][ktile][lane][16]
// ---------------------------------------------------------------------------
struct WDesc { const float* src; _Float16* dst; int K, N, Kpad, Npad; };
struct WTable { WDesc d[32]; int n; };

__global__ void cvt_weights_kernel(WTable t) {
  const int di = blockIdx.x;
  if (di >= t.n) return;
  const WDesc w = t.d[di];
  const int kTiles = w.Kpad >> 5;
  const int tot = w.Kpad * w.Npad;
  for (int e = threadIdx.x; e < tot; e += blockDim.x) {
    const int k = e / w.Npad, n = e % w.Npad;
    const float v = (k < w.K && n < w.N) ? w.src[(size_t)k * w.N + n] : 0.f;
    const int kb = k >> 5, kk = k & 31;
    const int h  = (kk >> 3) & 1;
    const int i  = (kk & 7) + ((kk >> 4) << 3);
    const int lane = (n & 15) + (h << 4);
    const int tn = n >> 4;
    w.dst[(((size_t)tn * kTiles + kb) * 32 + lane) * 16 + i] = (_Float16)v;
  }
}

// ---------------------------------------------------------------------------
// farthest point sampling over 1024 points (xyz = first 3 of 6 channels)
// ---------------------------------------------------------------------------
__global__ void fps_kernel(const float* __restrict__ obj, int npts, int* __restrict__ out)
{
  __shared__ float dist[kP];
  __shared__ float rv[256];
  __shared__ int   ri[256];
  __shared__ float cx, cy, cz;
  const float* base = obj + (size_t)blockIdx.x * kP * 6;
  const int tid = threadIdx.x;
  if (tid == 0) { out[blockIdx.x * npts] = 0; cx = base[0]; cy = base[1]; cz = base[2]; }
  __syncthreads();
  for (int p = tid; p < kP; p += blockDim.x) {
    const float dx = base[p*6]-cx, dy = base[p*6+1]-cy, dz = base[p*6+2]-cz;
    dist[p] = dx*dx + dy*dy + dz*dz;
  }
  __syncthreads();
  for (int m = 1; m < npts; ++m) {
    float best = -1.f; int bi = kP;
    for (int p = tid; p < kP; p += blockDim.x)
      if (dist[p] > best) { best = dist[p]; bi = p; }
    rv[tid] = best; ri[tid] = bi;
    __syncthreads();
    for (int s = blockDim.x >> 1; s > 0; s >>= 1) {
      if (tid < s) {
        if (rv[tid+s] > rv[tid] || (rv[tid+s] == rv[tid] && ri[tid+s] < ri[tid])) {
          rv[tid] = rv[tid+s]; ri[tid] = ri[tid+s];
        }
      }
      __syncthreads();
    }
    if (tid == 0) {
      const int cur = ri[0];
      out[blockIdx.x * npts + m] = cur;
      cx = base[cur*6]; cy = base[cur*6+1]; cz = base[cur*6+2];
    }
    __syncthreads();
    for (int p = tid; p < kP; p += blockDim.x) {
      const float dx = base[p*6]-cx, dy = base[p*6+1]-cy, dz = base[p*6+2]-cz;
      const float d = dx*dx + dy*dy + dz*dz;
      if (d < dist[p]) dist[p] = d;
    }
    __syncthreads();
  }
}

// FPS over 32 points (xyz1, stride 3), 16 picks, one small block
__global__ void fps2_kernel(const float* __restrict__ xyz1, int* __restrict__ out)
{
  __shared__ float dist[32];
  __shared__ float cc[3];
  const float* base = xyz1 + (size_t)blockIdx.x * 32 * 3;
  const int t = threadIdx.x;
  if (t == 0) { out[blockIdx.x * 16] = 0; cc[0]=base[0]; cc[1]=base[1]; cc[2]=base[2]; }
  __syncthreads();
  {
    const float dx = base[t*3]-cc[0], dy = base[t*3+1]-cc[1], dz = base[t*3+2]-cc[2];
    dist[t] = dx*dx + dy*dy + dz*dz;
  }
  __syncthreads();
  for (int m = 1; m < 16; ++m) {
    if (t == 0) {
      float bv = -1.f; int bi = 0;
      for (int p = 0; p < 32; ++p) if (dist[p] > bv) { bv = dist[p]; bi = p; }
      out[blockIdx.x * 16 + m] = bi;
      cc[0]=base[bi*3]; cc[1]=base[bi*3+1]; cc[2]=base[bi*3+2];
    }
    __syncthreads();
    const float dx = base[t*3]-cc[0], dy = base[t*3+1]-cc[1], dz = base[t*3+2]-cc[2];
    const float d = dx*dx + dy*dy + dz*dz;
    if (d < dist[t]) dist[t] = d;
    __syncthreads();
  }
}

__global__ void gather_xyz_kernel(const float* __restrict__ src, long long cloudStride,
                                  int elemStride, const int* __restrict__ idx,
                                  int nIdx, int idxStride, float* __restrict__ dst)
{
  const int c = blockIdx.x;
  for (int e = threadIdx.x; e < nIdx * 3; e += blockDim.x) {
    const int m = e / 3, d = e % 3;
    const int g = idx[c * idxStride + m];
    dst[((size_t)c * nIdx + m) * 3 + d] =
        src[(size_t)c * cloudStride + (size_t)g * elemStride + d];
  }
}

// ---------------------------------------------------------------------------
// ball query: ascending-index collection via wave32 ballot prefix compaction
// ---------------------------------------------------------------------------
__device__ int ball_collect(const float* __restrict__ pts, int stride, int nP,
                            float cx, float cy, float cz, float r2, int ns,
                            int* list, int lane)
{
  int count = 0;
  for (int b0 = 0; b0 < nP && count < ns; b0 += 32) {
    const int p = b0 + lane;
    bool pred = false;
    if (p < nP) {
      const float dx = pts[p*stride]   - cx;
      const float dy = pts[p*stride+1] - cy;
      const float dz = pts[p*stride+2] - cz;
      pred = (dx*dx + dy*dy + dz*dz) < r2;
    }
    const unsigned msk = (unsigned)__ballot(pred);
    if (pred) {
      const int pos = count + __popc(msk & ((1u << lane) - 1u));
      if (pos < ns) list[pos] = p;
    }
    count += __popc(msk);
  }
  return count < ns ? count : ns;
}

// pn_rel SA1 grouped rows: 8 centers x 8 samples, lda=32 (6 valid, pad zeroed)
__global__ void rel_rows_kernel(const float* __restrict__ obj, const int* __restrict__ fi,
                                float* __restrict__ rows, float* __restrict__ nxyz)
{
  __shared__ int lists[8][8];
  const int c = blockIdx.x;
  const int wid = threadIdx.x >> 5, lane = threadIdx.x & 31;
  const float* base = obj + (size_t)c * kP * 6;
  const int m = wid;
  const int ctr = fi[c * 32 + m];
  const float cx = base[ctr*6], cy = base[ctr*6+1], cz = base[ctr*6+2];
  const int cnt = ball_collect(base, 6, kP, cx, cy, cz, 0.16f, 8, lists[wid], lane);
  lds_fence();
  if (lane == 0) {
    const int first = cnt > 0 ? lists[wid][0] : 0;
    for (int s = cnt; s < 8; ++s) lists[wid][s] = first;
    nxyz[(c*8+m)*3+0] = cx; nxyz[(c*8+m)*3+1] = cy; nxyz[(c*8+m)*3+2] = cz;
  }
  lds_fence();
  for (int e = lane; e < 8 * 32; e += 32) {
    const int s = e / 32, k = e % 32;
    const int g = lists[wid][s];
    float v = 0.f;
    if (k < 3) v = base[g*6+k] - (k==0?cx:(k==1?cy:cz));
    rows[((size_t)(c*8+m)*8 + s)*32 + k] = v;
  }
}

// pn_obj SA1 grouped rows (chunked): 32 centers x 32 samples, lda=32 (6 valid)
__global__ void obj_sa1_rows_kernel(const float* __restrict__ obj, const int* __restrict__ fi,
                                    int c0, float* __restrict__ rows)
{
  __shared__ int lists[8][32];
  const int c = c0 + blockIdx.x;
  const int wid = threadIdx.x >> 5, lane = threadIdx.x & 31;
  const float* base = obj + (size_t)c * kP * 6;
  for (int m = wid; m < 32; m += 8) {
    const int ctr = fi[c * 32 + m];
    const float cx = base[ctr*6], cy = base[ctr*6+1], cz = base[ctr*6+2];
    const int cnt = ball_collect(base, 6, kP, cx, cy, cz, 0.04f, 32, lists[wid], lane);
    lds_fence();
    if (lane == 0) {
      const int first = cnt > 0 ? lists[wid][0] : 0;
      for (int s = cnt; s < 32; ++s) lists[wid][s] = first;
    }
    lds_fence();
    const size_t rbase = ((size_t)blockIdx.x * 32 + m) * 32;
    for (int e = lane; e < 32 * 32; e += 32) {
      const int s = e / 32, k = e % 32;
      const int g = lists[wid][s];
      float v = 0.f;
      if (k < 3)      v = base[g*6+k] - (k==0?cx:(k==1?cy:cz));
      else if (k < 6) v = base[g*6+k];
      rows[(rbase + s) * 32 + k] = v;
    }
    lds_fence();
  }
}

// pn_obj SA2 grouped rows (chunked): 16 centers x 32 samples, lda=160 (131 valid)
__global__ void obj_sa2_rows_kernel(const float* __restrict__ xyz1, const float* __restrict__ f1,
                                    const int* __restrict__ fi2, int c0, float* __restrict__ rows)
{
  __shared__ int lists[8][32];
  const int c = c0 + blockIdx.x;
  const int wid = threadIdx.x >> 5, lane = threadIdx.x & 31;
  const float* pts = xyz1 + (size_t)c * 32 * 3;
  const float* fts = f1   + (size_t)c * 32 * kD;
  for (int m = wid; m < 16; m += 8) {
    const int ctr = fi2[c * 16 + m];
    const float cx = pts[ctr*3], cy = pts[ctr*3+1], cz = pts[ctr*3+2];
    const int cnt = ball_collect(pts, 3, 32, cx, cy, cz, 0.16f, 32, lists[wid], lane);
    lds_fence();
    if (lane == 0) {
      const int first = cnt > 0 ? lists[wid][0] : 0;
      for (int s = cnt; s < 32; ++s) lists[wid][s] = first;
    }
    lds_fence();
    const size_t rbase = ((size_t)blockIdx.x * 16 + m) * 32;
    for (int e = lane; e < 32 * 160; e += 32) {
      const int s = e / 160, k = e % 160;
      const int g = lists[wid][s];
      float v = 0.f;
      if (k < 3)        v = pts[g*3+k] - (k==0?cx:(k==1?cy:cz));
      else if (k < 131) v = fts[(size_t)g * kD + (k - 3)];
      rows[(rbase + s) * 160 + k] = v;
    }
    lds_fence();
  }
}

// out[r] = concat(xyz[r](3), feat[r](F)), zero-padded to ldout
__global__ void concat_kernel(const float* __restrict__ xyz, const float* __restrict__ feat,
                              int F, float* __restrict__ out, int ldout, int R)
{
  const int r = blockIdx.x;
  if (r >= R) return;
  for (int k = threadIdx.x; k < ldout; k += blockDim.x) {
    float v = 0.f;
    if (k < 3)          v = xyz[(size_t)r*3 + k];
    else if (k < 3 + F) v = feat[(size_t)r*F + (k - 3)];
    out[(size_t)r * ldout + k] = v;
  }
}

// out[g,c] = max_s in[g*S+s, c]; optional second (strided, col-offset) copy
__global__ void maxpool_kernel(const float* __restrict__ in, float* __restrict__ out,
                               int S, int C, int ldin,
                               float* __restrict__ out2, int ldo2, int colOff)
{
  const size_t g = blockIdx.x;
  for (int c = threadIdx.x; c < C; c += blockDim.x) {
    float mx = in[(g * S) * (size_t)ldin + c];
    for (int s = 1; s < S; ++s) {
      const float v = in[(g * S + s) * (size_t)ldin + c];
      mx = v > mx ? v : mx;
    }
    out[g * (size_t)C + c] = mx;
    if (out2) out2[g * (size_t)ldo2 + colOff + c] = mx;
  }
}

// xe[b*24+t] = emb[lang[b,t]]  (fp32 gather)
__global__ void lang_gather_kernel(const int* __restrict__ lang, const float* __restrict__ emb,
                                   float* __restrict__ xe)
{
  const int r = blockIdx.x;           // b*24+t
  const int tok = lang[r];
  for (int k = threadIdx.x; k < 64; k += blockDim.x)
    xe[(size_t)r * 64 + k] = emb[(size_t)tok * 64 + k];
}

// ---------------------------------------------------------------------------
// fused LSTM (single block, 8 waves). z = xt@Wi + h@Wh + b via WMMA, then
// gates + masked running max of h over time. Wi/Wh in packed fragment layout.
// ---------------------------------------------------------------------------
__global__ __launch_bounds__(256) void lstm_kernel(
    const float* __restrict__ xp, const int* __restrict__ lang,
    const _Float16* __restrict__ wi, const _Float16* __restrict__ wh,
    const float* __restrict__ bias, float* __restrict__ lf)
{
  __shared__ float hls[16 * 128];   // rows 8..15 are don't-care lanes
  __shared__ float cls[kB * 128], hmax[kB * 128], zls[kB * 512];
  const int tid = threadIdx.x, lane = tid & 31, wid = tid >> 5;
  const int half = lane >> 4, l16 = lane & 15;
  for (int e = tid; e < 16 * 128; e += blockDim.x) hls[e] = 0.f;
  for (int e = tid; e < kB * 128; e += blockDim.x) { cls[e] = 0.f; hmax[e] = -1e9f; }
  __syncthreads();
  const float* xrowBase = xp + (size_t)l16 * kL * 64;
  for (int t = 0; t < kL; ++t) {
    const float* xrow = xrowBase + (size_t)t * 64;
    for (int tn = wid; tn < 32; tn += 8) {
      const int colBase = tn << 4;
      v8f acc = {};
      const _Float16* wip = wi + ((size_t)tn * 2 * 32 + lane) * 16;
      for (int kb = 0; kb < 2; ++kb) {
        const int base = (kb << 5) + (half << 3);
        const v16h a = cvt_frag<false>(xrow + base, xrow + base + 16);
        const v16h b = *(const v16h*)(wip + (size_t)kb * 512);
        acc = __builtin_amdgcn_wmma_f32_16x16x32_f16(false, a, false, b, (short)0, acc, false, false);
      }
      const _Float16* whp = wh + ((size_t)tn * 4 * 32 + lane) * 16;
      const float* hrowBase = hls + l16 * 128;
      for (int kb = 0; kb < 4; ++kb) {
        const int base = (kb << 5) + (half << 3);
        const v16h a = cvt_frag<false>(hrowBase + base, hrowBase + base + 16);
        const v16h b = *(const v16h*)(whp + (size_t)kb * 512);
        acc = __builtin_amdgcn_wmma_f32_16x16x32_f16(false, a, false, b, (short)0, acc, false, false);
      }
#pragma unroll
      for (int r = 0; r < 8; ++r) {
        const int rr = r + (half << 3);
        const int cc = colBase + l16;
        if (rr < kB) zls[rr * 512 + cc] = acc[r] + bias[cc];
      }
    }
    __syncthreads();
    for (int e = tid; e < kB * 128; e += blockDim.x) {
      const int r = e >> 7, c = e & 127;
      const float iv = zls[r*512 + c],       fv = zls[r*512 + 128 + c];
      const float gv = zls[r*512 + 256 + c], ov = zls[r*512 + 384 + c];
      const float cn = sigm(fv) * cls[e] + sigm(iv) * tanhf(gv);
      const float hn = sigm(ov) * tanhf(cn);
      cls[e] = cn; hls[r * 128 + c] = hn;
      const float cand = (lang[r * kL + t] != 0) ? hn : -1e9f;
      if (cand > hmax[e]) hmax[e] = cand;
    }
    __syncthreads();
  }
  for (int e = tid; e < kB * 128; e += blockDim.x) lf[e] = hmax[e];
}

// ---------------------------------------------------------------------------
// DGCNN helpers
// ---------------------------------------------------------------------------
__global__ void gin_kernel(const float* __restrict__ relpn, const float* __restrict__ lf,
                           float* __restrict__ gin)
{
  const int r = blockIdx.x;        // b*52+n
  const int bb = r / kN;
  for (int k = threadIdx.x; k < 128; k += blockDim.x) {
    gin[(size_t)r * 256 + k]       = relpn[(size_t)r * 128 + k];
    gin[(size_t)r * 256 + 128 + k] = lf[bb * 128 + k];
  }
}

__global__ void knn_kernel(const float* __restrict__ h, int C, int* __restrict__ idx)
{
  __shared__ float d2s[kN * kN];
  const int bb = blockIdx.x;
  for (int e = threadIdx.x; e < kN * kN; e += blockDim.x) {
    const int i = e / kN, j = e % kN;
    const float* hi = h + (size_t)(bb * kN + i) * C;
    const float* hj = h + (size_t)(bb * kN + j) * C;
    float s = 0.f;
    for (int k = 0; k < C; ++k) { const float d = hi[k] - hj[k]; s += d * d; }
    d2s[e] = s;
  }
  __syncthreads();
  for (int i = threadIdx.x; i < kN; i += blockDim.x) {
    int s0 = 0, s1 = 0, s2 = 0;
    float bv = 3.4e38f;
    for (int j = 0; j < kN; ++j) if (d2s[i*kN+j] < bv) { bv = d2s[i*kN+j]; s0 = j; }
    bv = 3.4e38f;
    for (int j = 0; j < kN; ++j) if (j != s0 && d2s[i*kN+j] < bv) { bv = d2s[i*kN+j]; s1 = j; }
    bv = 3.4e38f;
    for (int j = 0; j < kN; ++j) if (j != s0 && j != s1 && d2s[i*kN+j] < bv) { bv = d2s[i*kN+j]; s2 = j; }
    idx[(bb*kN+i)*3+0] = s0; idx[(bb*kN+i)*3+1] = s1; idx[(bb*kN+i)*3+2] = s2;
  }
}

__global__ void edge_rows_kernel(const float* __restrict__ h, int C, const int* __restrict__ idx,
                                 float* __restrict__ e)
{
  const int gi = blockIdx.x;          // b*kN + i
  const int bb = gi / kN;
  const float* hi = h + (size_t)gi * C;
  for (int t = threadIdx.x; t < 3 * 2 * C; t += blockDim.x) {
    const int s = t / (2 * C), k = t % (2 * C);
    const int j = idx[gi * 3 + s];
    const float* hj = h + (size_t)(bb * kN + j) * C;
    const float v = (k < C) ? hi[k] : (hj[k - C] - hi[k - C]);
    e[((size_t)gi * 3 + s) * (2 * C) + k] = v;
  }
}

__global__ void obj_transpose_kernel(const float* __restrict__ t, float* __restrict__ out)
{
  const int r = blockIdx.x;          // b*kN + n
  const int bb = r / kN, n = r % kN;
  for (int c = threadIdx.x; c < 607; c += blockDim.x)
    out[((size_t)n * kB + bb) * 607 + c] = t[(size_t)r * 607 + c];
}

// ---------------------------------------------------------------------------
// host side
// ---------------------------------------------------------------------------
enum {
  W_PRP_G0, W_PRP_G1, W_PRP_G2, W_PRP_S0, W_PRP_S1, W_PRP_S2,
  W_POP_G0, W_POP_G1, W_POP_G2, W_POP_S10, W_POP_S11, W_POP_S12,
  W_POP_S20, W_POP_S21, W_POP_S22,
  W_DG_E0, W_DG_E1, W_DG_E2, W_DG_E3, W_DG_FIN,
  W_LPROJ, W_LWH, W_LWI,
  W_MLP0, W_MLP1, W_MLP2,
  W_REL1, W_REL2, W_MREL1, W_MREL2,
  W_COUNT
};

extern "C" void kernel_launch(void* const* d_in, const int* in_sizes, int n_in,
                              void* d_out, int out_size, void* d_ws, size_t ws_size,
                              hipStream_t stream)
{
  (void)in_sizes; (void)n_in; (void)out_size; (void)ws_size;

  const float* objects = (const float*)d_in[1];
  const int*   lang    = (const int*)d_in[3];
  auto bin = [&](int i) { return (const float*)d_in[i]; };

  // weight list: {d_in index, K, N} in enum order (jax pytree: dict keys sorted)
  struct WInfoH { int idx, K, N; };
  static const WInfoH winf[W_COUNT] = {
    {5,67,64},{7,64,64},{9,64,128},{11,6,32},{13,32,32},{15,32,64},
    {17,259,256},{19,256,512},{21,512,128},{23,6,64},{25,64,64},{27,64,128},
    {29,131,128},{31,128,128},{33,128,256},
    {35,512,128},{37,256,128},{39,256,128},{41,256,128},{43,512,128},
    {47,64,64},{48,128,512},{49,64,512},
    {51,128,128},{53,128,256},{55,256,607},
    {57,256,128},{59,128,128},{61,128,128},{63,128,128},
  };

  char* ws = (char*)d_ws;
  size_t off = 0;
  auto alloc = [&](size_t bytes) -> void* {
    void* p = ws + off;
    off = (off + bytes + 255) & ~(size_t)255;
    return p;
  };

  const _Float16* wptr[W_COUNT];
  WTable table; table.n = W_COUNT;
  for (int k = 0; k < W_COUNT; ++k) {
    const int Kp = (winf[k].K + 31) & ~31, Np = (winf[k].N + 15) & ~15;
    _Float16* dst = (_Float16*)alloc((size_t)Kp * Np * sizeof(_Float16));
    WDesc d; d.src = bin(winf[k].idx); d.dst = dst;
    d.K = winf[k].K; d.N = winf[k].N; d.Kpad = Kp; d.Npad = Np;
    table.d[k] = d;
    wptr[k] = dst;
  }

  int*   fi      = (int*)  alloc((size_t)kNC * 32 * 4);
  int*   fi2     = (int*)  alloc((size_t)kNC * 16 * 4);
  float* xyz1    = (float*)alloc((size_t)kNC * 32 * 3 * 4);
  float* xyz2    = (float*)alloc((size_t)kNC * 16 * 3 * 4);
  float* nxyzrel = (float*)alloc((size_t)kNC * 8 * 3 * 4);
  float* bufA    = (float*)alloc((size_t)2200064 * 4);
  float* bufB    = (float*)alloc((size_t)3407936 * 4);
  float* bufC    = (float*)alloc((size_t)3407936 * 4);
  float* f1rel   = (float*)alloc((size_t)kNC * 8 * 64 * 4);
  float* relpn   = (float*)alloc((size_t)kNC * 128 * 4);
  float* f1obj   = (float*)alloc((size_t)kNC * 32 * 128 * 4);
  float* f2obj   = (float*)alloc((size_t)kNC * 16 * 256 * 4);
  float* ofbuf   = (float*)alloc((size_t)kNC * 128 * 4);
  float* xe      = (float*)alloc((size_t)192 * 64 * 4);
  float* xpb     = (float*)alloc((size_t)16 * kL * 64 * 4);  // rows 8..15 don't-care
  float* lfb     = (float*)alloc((size_t)kB * 128 * 4);
  float* hbuf    = (float*)alloc((size_t)kNC * 256 * 4);
  float* hn0     = (float*)alloc((size_t)kNC * 128 * 4);
  float* hn1     = (float*)alloc((size_t)kNC * 128 * 4);
  float* catb    = (float*)alloc((size_t)kNC * 512 * 4);
  float* gbuf    = (float*)alloc((size_t)kNC * 128 * 4);
  float* tbuf    = (float*)alloc((size_t)kB * kN * 3 * 128 * 4);
  int*   knnIdx  = (int*)  alloc((size_t)kB * kN * 3 * 4);
  float* r1      = (float*)alloc((size_t)21632 * 128 * 4);
  float* objtmp  = (float*)alloc((size_t)kNC * 608 * 4);

  auto gemm = [&](int amode, const float* A, int lda, int w, const float* bias,
                  float* Cp, int ldc, int M, int N, int K, int act, const float* G) {
    const int tiles = (M / 16) * ((N + 15) / 16);
    int grid = (tiles + 7) / 8;
    if (grid > 8192) grid = 8192;
    if (grid < 1) grid = 1;
    if (amode == 0)
      gemm_kernel<0><<<grid,256,0,stream>>>(A,lda,wptr[w],bias,Cp,ldc,M,N,K,act,G);
    else if (amode == 1)
      gemm_kernel<1><<<grid,256,0,stream>>>(A,lda,wptr[w],bias,Cp,ldc,M,N,K,act,G);
    else
      gemm_kernel<2><<<grid,256,0,stream>>>(A,lda,wptr[w],bias,Cp,ldc,M,N,K,act,G);
  };

  // 0) weights -> f16 packed fragments
  cvt_weights_kernel<<<W_COUNT,256,0,stream>>>(table);

  // 1) sampling (shared between pn_rel and pn_obj: FPS prefix property)
  fps_kernel<<<kNC,256,0,stream>>>(objects, 32, fi);
  gather_xyz_kernel<<<kNC,128,0,stream>>>(objects, (long long)kP*6, 6, fi, 32, 32, xyz1);
  fps2_kernel<<<kNC,32,0,stream>>>(xyz1, fi2);
  gather_xyz_kernel<<<kNC,64,0,stream>>>(xyz1, (long long)32*3, 3, fi2, 16, 16, xyz2);

  // 2) pn_rel (masked features): SA1 8x8 then glob
  rel_rows_kernel<<<kNC,256,0,stream>>>(objects, fi, bufA, nxyzrel);
  gemm(0, bufA, 32, W_PRP_S0, bin(10), bufB, 32, 26624, 32, 6,  1, nullptr);
  gemm(0, bufB, 32, W_PRP_S1, bin(12), bufC, 32, 26624, 32, 32, 1, nullptr);
  gemm(0, bufC, 32, W_PRP_S2, bin(14), bufB, 64, 26624, 64, 32, 1, nullptr);
  maxpool_kernel<<<3328,64,0,stream>>>(bufB, f1rel, 8, 64, 64, nullptr, 0, 0);
  concat_kernel<<<3328,96,0,stream>>>(nxyzrel, f1rel, 64, bufA, 96, 3328);
  gemm(0, bufA, 96, W_PRP_G0, bin(4), bufB, 64, 3328, 64, 67, 1, nullptr);
  gemm(0, bufB, 64, W_PRP_G1, bin(6), bufC, 64, 3328, 64, 64, 1, nullptr);
  gemm(0, bufC, 64, W_PRP_G2, bin(8), bufB, 128, 3328, 128, 64, 1, nullptr);
  maxpool_kernel<<<kNC,128,0,stream>>>(bufB, relpn, 8, 128, 128, nullptr, 0, 0);

  // 3) pn_obj SA1 (chunks of 26 clouds -> 26624 rows)
  for (int c0 = 0; c0 < kNC; c0 += 26) {
    obj_sa1_rows_kernel<<<26,256,0,stream>>>(objects, fi, c0, bufA);
    gemm(0, bufA, 32, W_POP_S10, bin(22), bufB, 64,  26624, 64,  6,  1, nullptr);
    gemm(0, bufB, 64, W_POP_S11, bin(24), bufC, 64,  26624, 64,  64, 1, nullptr);
    gemm(0, bufC, 64, W_POP_S12, bin(26), bufB, 128, 26624, 128, 64, 1, nullptr);
    maxpool_kernel<<<832,128,0,stream>>>(bufB, f1obj + (size_t)c0*32*128, 32, 128, 128, nullptr, 0, 0);
  }
  // 4) pn_obj SA2 (chunks of 26 clouds -> 13312 rows, K=131 pad 160)
  for (int c0 = 0; c0 < kNC; c0 += 26) {
    obj_sa2_rows_kernel<<<26,256,0,stream>>>(xyz1, f1obj, fi2, c0, bufA);
    gemm(0, bufA, 160, W_POP_S20, bin(28), bufB, 128, 13312, 128, 131, 1, nullptr);
    gemm(0, bufB, 128, W_POP_S21, bin(30), bufC, 128, 13312, 128, 128, 1, nullptr);
    gemm(0, bufC, 128, W_POP_S22, bin(32), bufB, 256, 13312, 256, 128, 1, nullptr);
    maxpool_kernel<<<416,256,0,stream>>>(bufB, f2obj + (size_t)c0*16*256, 32, 256, 256, nullptr, 0, 0);
  }
  // 5) pn_obj glob (K=259 pad 288)
  concat_kernel<<<6656,256,0,stream>>>(xyz2, f2obj, 256, bufA, 288, 6656);
  gemm(0, bufA, 288, W_POP_G0, bin(16), bufB, 256, 6656, 256, 259, 1, nullptr);
  gemm(0, bufB, 256, W_POP_G1, bin(18), bufC, 512, 6656, 512, 256, 1, nullptr);
  gemm(0, bufC, 512, W_POP_G2, bin(20), bufB, 128, 6656, 128, 512, 1, nullptr);
  maxpool_kernel<<<kNC,128,0,stream>>>(bufB, ofbuf, 16, 128, 128, nullptr, 0, 0);

  // 6) language LSTM
  lang_gather_kernel<<<192,64,0,stream>>>(lang, bin(45), xe);
  gemm(0, xe, 64, W_LPROJ, bin(46), xpb, 64, 192, 64, 64, 1, nullptr);
  lstm_kernel<<<1,256,0,stream>>>(xpb, lang, wptr[W_LWI], wptr[W_LWH], bin(44), lfb);

  // 7) DGCNN
  gin_kernel<<<kNC,128,0,stream>>>(relpn, lfb, hbuf);
  {
    const float* curH = hbuf; int Cc = 256;
    float* hnArr[2] = { hn0, hn1 };
    const int eW[4] = { W_DG_E0, W_DG_E1, W_DG_E2, W_DG_E3 };
    const int eB[4] = { 34, 36, 38, 40 };
    for (int l = 0; l < 4; ++l) {
      knn_kernel<<<kB,256,0,stream>>>(curH, Cc, knnIdx);
      edge_rows_kernel<<<kNC,256,0,stream>>>(curH, Cc, knnIdx, bufA);
      gemm(0, bufA, 2*Cc, eW[l], bin(eB[l]), tbuf, 128, kB*kN*3, 128, 2*Cc, 2, nullptr);
      maxpool_kernel<<<kNC,128,0,stream>>>(tbuf, hnArr[l & 1], 3, 128, 128, catb, 512, l*128);
      curH = hnArr[l & 1]; Cc = 128;
    }
    gemm(0, catb, 512, W_DG_FIN, bin(42), gbuf, 128, kNC, 128, 512, 2, nullptr);
  }

  // 8) rel / mrel heads (u never materialized: AMODE==2 pair loader)
  float* out     = (float*)d_out;
  float* relOut  = out + 252512;              // 52*8*607
  float* mrelOut = out + 252512 + 2768896;    // + 52*52*8*128
  gemm(2, nullptr, 0,   W_REL1,  bin(56), r1,      128, 21632, 128, 256, 0, gbuf);
  gemm(1, r1,     128,  W_REL2,  bin(58), relOut,  128, 21632, 128, 128, 0, nullptr);
  gemm(1, relOut, 128,  W_MREL1, bin(60), r1,      128, 21632, 128, 128, 0, nullptr);
  gemm(1, r1,     128,  W_MREL2, bin(62), mrelOut, 128, 21632, 128, 128, 0, nullptr);

  // 9) obj head
  gemm(0, ofbuf, 128, W_MLP0, bin(50), bufB,   128, kNC, 128, 128, 1, nullptr);
  gemm(0, bufB,  128, W_MLP1, bin(52), bufC,   256, kNC, 256, 128, 1, nullptr);
  gemm(0, bufC,  256, W_MLP2, bin(54), objtmp, 607, kNC, 607, 256, 0, nullptr);
  obj_transpose_kernel<<<kNC,256,0,stream>>>(objtmp, out);
}
